// HNO_6270652252904
// MI455X (gfx1250) — matrix operand order repaired and checked by reading, not compiled, not run on hardware
//
#include <hip/hip_runtime.h>
#include <hip/hip_bf16.h>
#include <math.h>

// ---------------- problem constants (from reference) ----------------
#define H       128
#define WPOOL   16
#define NRES    2191
#define NMAIN   6573
#define NBACK   2191
#define BG      8
#define NPER    (NMAIN + NBACK)            // 8764
#define NTOT    (BG * NPER)                // 70112
#define EDG     (NTOT * 16)                // 1121792
#define EREF    (NRES * 16)                // 35056
#define NINF    4224                       // 2*WPOOL*H + H
#define MROWS   (BG * NRES)                // 17528
#define EPSF    1e-5f
#define BN_EVALF 0.99999500003749968750f   // 1/sqrt(1+1e-5)

typedef _Float16 half_t;
typedef half_t v16h __attribute__((ext_vector_type(16)));
typedef half_t v8h  __attribute__((ext_vector_type(8)));
typedef float  v8f  __attribute__((ext_vector_type(8)));
typedef int    v4i  __attribute__((ext_vector_type(4)));

// ---------------- CDNA5 async global->LDS machinery (guarded) ----------------
#if defined(__HIP_DEVICE_COMPILE__) && __has_builtin(__builtin_amdgcn_global_load_async_to_lds_b128)
#define GEMM_ASYNC 1
#endif

#ifdef GEMM_ASYNC
#if __has_builtin(__builtin_amdgcn_s_wait_asynccnt)
#define WAIT_ASYNC() __builtin_amdgcn_s_wait_asynccnt(0)
#else
#define WAIT_ASYNC() asm volatile("s_wait_asynccnt 0" ::: "memory")
#endif
#else
#define WAIT_ASYNC()
#endif

__device__ __forceinline__ void async_b128(const half_t* g, half_t* l) {
#ifdef GEMM_ASYNC
    // global (AS1) -> LDS (AS3); pointee type is int4 (b128 per lane).
    // LDS offset = low 32 bits of the flat shared-aperture address.
    __builtin_amdgcn_global_load_async_to_lds_b128(
        (__attribute__((address_space(1))) v4i*)(uintptr_t)g,
        (__attribute__((address_space(3))) v4i*)(unsigned)(uintptr_t)l,
        0, 0);
#else
    (void)g; (void)l;
#endif
}

// ---------------- small utility kernels ----------------
__global__ void k_zero(float* __restrict__ p, long long n) {
    long long i = (long long)blockIdx.x * blockDim.x + threadIdx.x;
    if (i < n) p[i] = 0.f;
}

__global__ void k_deg(const int* __restrict__ src, float* __restrict__ deg, int ne) {
    int e = blockIdx.x * blockDim.x + threadIdx.x;
    if (e < ne) atomicAdd(&deg[src[e]], 1.f);
}

__global__ void k_norm(const int* __restrict__ src, const int* __restrict__ dst,
                       const float* __restrict__ deg, float* __restrict__ nrm, int ne) {
    int e = blockIdx.x * blockDim.x + threadIdx.x;
    if (e >= ne) return;
    float a = deg[src[e]], b = deg[dst[e]];
    float ia = (a > 0.f) ? rsqrtf(fmaxf(a, 1e-12f)) : 0.f;
    float ib = (b > 0.f) ? rsqrtf(fmaxf(b, 1e-12f)) : 0.f;
    nrm[e] = -(ia * ib);
}

// out[dst] += norm[e] * t[src]  (per feature)
__global__ void k_prop(const int* __restrict__ src, const int* __restrict__ dst,
                       const float* __restrict__ nrm, const float* __restrict__ t,
                       float* __restrict__ out, int ne, int fin) {
    long long i = (long long)blockIdx.x * blockDim.x + threadIdx.x;
    if (i >= (long long)ne * fin) return;
    int e = (int)(i / fin);
    int c = (int)(i % fin);
    atomicAdd(&out[(long long)dst[e] * fin + c], nrm[e] * t[(long long)src[e] * fin + c]);
}

// t2 = 2*t2 - t0
__global__ void k_t2(float* __restrict__ t2, const float* __restrict__ t0, long long n) {
    long long i = (long long)blockIdx.x * blockDim.x + threadIdx.x;
    if (i < n) t2[i] = 2.f * t2[i] - t0[i];
}

// act = tanh( b + T0@W[0] + T1@W[1] + T2@W[2] ),  W layout (3, fin, H) contiguous
__global__ void k_cheb(const float* __restrict__ t0, const float* __restrict__ t1,
                       const float* __restrict__ t2, const float* __restrict__ W,
                       const float* __restrict__ bias, float* __restrict__ act,
                       int n, int fin) {
    long long i = (long long)blockIdx.x * blockDim.x + threadIdx.x;
    if (i >= (long long)n * H) return;
    int row = (int)(i / H), col = (int)(i % H);
    const float* w0 = W;
    const float* w1 = W + (long long)fin * H;
    const float* w2 = W + 2ll * fin * H;
    const float* p0 = t0 + (long long)row * fin;
    const float* p1 = t1 + (long long)row * fin;
    const float* p2 = t2 + (long long)row * fin;
    float acc = bias[col];
    for (int f = 0; f < fin; ++f)
        acc += p0[f] * w0[f * H + col] + p1[f] * w1[f * H + col] + p2[f] * w2[f * H + col];
    act[i] = tanhf(acc);
}

// per-column mean / rstd over n rows (biased var, as torch BN training)
__global__ void k_bn_reduce(const float* __restrict__ act, int n,
                            float* __restrict__ mean, float* __restrict__ rstd) {
    int col = blockIdx.x;  // 0..H-1
    __shared__ float ssum[256], ssq[256];
    float s = 0.f, q = 0.f;
    for (int r = threadIdx.x; r < n; r += blockDim.x) {
        float v = act[(long long)r * H + col];
        s += v; q += v * v;
    }
    ssum[threadIdx.x] = s; ssq[threadIdx.x] = q;
    __syncthreads();
    for (int o = 128; o > 0; o >>= 1) {
        if ((int)threadIdx.x < o) {
            ssum[threadIdx.x] += ssum[threadIdx.x + o];
            ssq[threadIdx.x]  += ssq[threadIdx.x + o];
        }
        __syncthreads();
    }
    if (threadIdx.x == 0) {
        float m = ssum[0] / (float)n;
        float var = ssq[0] / (float)n - m * m;
        mean[col] = m;
        rstd[col] = rsqrtf(var + EPSF);
    }
}

__global__ void k_bn_apply(const float* __restrict__ act, const float* __restrict__ mean,
                           const float* __restrict__ rstd, float* __restrict__ h, long long n) {
    long long i = (long long)blockIdx.x * blockDim.x + threadIdx.x;
    if (i >= n) return;
    int col = (int)(i % H);
    h[i] = (act[i] - mean[col]) * rstd[col];
}

// adaptive average pool over gathered rows; writes both `full` and the d_out slice
__global__ void k_pool(const float* __restrict__ h, const int* __restrict__ idx, int nm,
                       float* __restrict__ full, int full_off, float* __restrict__ out_part) {
    int t = blockIdx.x * blockDim.x + threadIdx.x;
    if (t >= BG * WPOOL * H) return;
    int c = t % H;
    int w = (t / H) % WPOOL;
    int b = t / (H * WPOOL);
    int start = (w * nm) / WPOOL;
    int end   = ((w + 1) * nm + WPOOL - 1) / WPOOL;
    float s = 0.f;
    for (int i = start; i < end; ++i)
        s += h[(long long)idx[b * nm + i] * H + c];
    float v = s / (float)(end - start);
    full[b * (2 * WPOOL * H) + full_off + w * H + c] = v;
    out_part[(b * WPOOL + w) * H + c] = v;
}

// weight prep: f32 (K x N) -> f16 transposed (N x K), so GEMM B-tiles are K-contiguous
__global__ void k_w_f16_t(const float* __restrict__ in, half_t* __restrict__ out,
                          int Kd, int N) {
    long long i = (long long)blockIdx.x * blockDim.x + threadIdx.x;
    if (i >= (long long)Kd * N) return;
    int k = (int)(i / N), n = (int)(i % N);
    out[(size_t)n * Kd + k] = (half_t)in[i];
}

// res[b*NRES+j, :] = concat(hr[j, 0:H], full[b, 0:4096]) -> f16
__global__ void k_build_res(const float* __restrict__ hr, const float* __restrict__ full,
                            half_t* __restrict__ res) {
    long long i = (long long)blockIdx.x * blockDim.x + threadIdx.x;
    if (i >= (long long)MROWS * NINF) return;
    int row = (int)(i / NINF), col = (int)(i % NINF);
    int b = row / NRES, j = row % NRES;
    float v = (col < H) ? hr[(long long)j * H + col]
                        : full[b * (2 * WPOOL * H) + (col - H)];
    res[i] = (half_t)v;
}

// ---------------- WMMA GEMM: C = epilogue(A[MxK] * Bt[NxK]^T) ----------------
// 128x128 block tile, 256 threads = 8 waves (2m x 4n), each wave 64x32 = 4x2 WMMAs.
// Double-buffered LDS; tiles staged with async global->LDS DMA when available.
#define BM 128
#define BN 128
#define BKK 32
#define GT  256

__global__ __launch_bounds__(GT)
void k_gemm_wmma(const half_t* __restrict__ A, const half_t* __restrict__ Bt,
                 half_t* __restrict__ C, int M, int N, int Kd, int relu_bn) {
    __shared__ __attribute__((aligned(64))) half_t As[2][BM][BKK];  // [m][k]
    __shared__ __attribute__((aligned(64))) half_t Bs[2][BN][BKK];  // [n][k]

    const int tid   = threadIdx.x;
    const int lane  = tid & 31;
    const int wave  = tid >> 5;
    const int wm    = wave >> 2;       // 0..1  (64 rows each)
    const int wn    = wave & 3;        // 0..3  (32 cols each)
    const int lhalf = lane >> 4;       // 0 / 1
    const int l15   = lane & 15;
    const int bM = blockIdx.y * BM;
    const int bN = blockIdx.x * BN;

    v8f acc[4][2] = {};

    // one K-step tile stage: 512 b128 segments each for A and B, 2+2 per thread
    auto stage = [&](int buf, int k0) {
        #pragma unroll
        for (int s = 0; s < 2; ++s) {
            int seg = tid + s * GT;        // 0..511
            int row = seg >> 2;            // 0..127
            int col = (seg & 3) * 8;       // 0/8/16/24
            int gr  = bM + row;
#ifdef GEMM_ASYNC
            if (gr < M) {
                async_b128(A + (size_t)gr * Kd + k0 + col, &As[buf][row][col]);
            } else {
                v8h z = {};
                *(v8h*)&As[buf][row][col] = z;
            }
            async_b128(Bt + (size_t)(bN + row) * Kd + k0 + col, &Bs[buf][row][col]);
#else
            v8h va = {};
            if (gr < M) va = *(const v8h*)(A + (size_t)gr * Kd + k0 + col);
            *(v8h*)&As[buf][row][col] = va;
            *(v8h*)&Bs[buf][row][col] = *(const v8h*)(Bt + (size_t)(bN + row) * Kd + k0 + col);
            if (k0 + BKK < Kd) {
                if (gr < M) __builtin_prefetch(A + (size_t)gr * Kd + k0 + BKK + col, 0, 3);
                __builtin_prefetch(Bt + (size_t)(bN + row) * Kd + k0 + BKK + col, 0, 3);
            }
#endif
        }
    };

    const int nk = Kd / BKK;
    stage(0, 0);
    WAIT_ASYNC();
    __syncthreads();

    for (int kt = 0; kt < nk; ++kt) {
        int cur = kt & 1;
        if (kt + 1 < nk) stage(cur ^ 1, (kt + 1) * BKK);  // DMA next tile during compute

        // A fragments: lane half selects K base {0,16};
        // elements 0..15 -> K base+0..15 (matches B)
        v16h af[4], bf[2];
        #pragma unroll
        for (int mt = 0; mt < 4; ++mt) {
            int row  = wm * 64 + mt * 16 + l15;
            int base = lhalf * 16;
            v8h lo = *(const v8h*)&As[cur][row][base];
            v8h hi = *(const v8h*)&As[cur][row][base + 8];
            v16h a;
            #pragma unroll
            for (int j = 0; j < 8; ++j) { a[j] = lo[j]; a[8 + j] = hi[j]; }
            af[mt] = a;
        }
        // B fragments: element i -> K = lhalf*16 + i, col = lane%16 (Bs is [n][k])
        #pragma unroll
        for (int nt = 0; nt < 2; ++nt) {
            int coln = wn * 32 + nt * 16 + l15;
            bf[nt] = *(const v16h*)&Bs[cur][coln][lhalf * 16];
        }
        #pragma unroll
        for (int mt = 0; mt < 4; ++mt)
            #pragma unroll
            for (int nt = 0; nt < 2; ++nt)
                acc[mt][nt] = __builtin_amdgcn_wmma_f32_16x16x32_f16(
                    false, af[mt], false, bf[nt],
                    (short)0, acc[mt][nt], false, false);

        WAIT_ASYNC();
        __syncthreads();
    }

    // epilogue: D VGPR j -> M = j + (lane/16)*8, N = lane%16
    #pragma unroll
    for (int mt = 0; mt < 4; ++mt)
        #pragma unroll
        for (int nt = 0; nt < 2; ++nt)
            #pragma unroll
            for (int j = 0; j < 8; ++j) {
                int m  = bM + wm * 64 + mt * 16 + lhalf * 8 + j;
                int nn = bN + wn * 32 + nt * 16 + l15;
                if (m < M) {
                    float v = acc[mt][nt][j];
                    if (relu_bn) { v *= BN_EVALF; v = fmaxf(v, 0.f); }
                    C[(size_t)m * N + nn] = (half_t)v;
                }
            }
}

// final projection: out[row,c] = sum_k r2[row,k]*W2[k,c] + b2[c]
__global__ void k_gemm3(const half_t* __restrict__ r2, const float* __restrict__ W2,
                        const float* __restrict__ b2, float* __restrict__ out) {
    int t = blockIdx.x * blockDim.x + threadIdx.x;
    if (t >= MROWS * 3) return;
    int row = t / 3, c = t % 3;
    const half_t* rp = r2 + (size_t)row * NINF;
    float acc = b2[c];
    for (int k = 0; k < NINF; ++k)
        acc += (float)rp[k] * W2[k * 3 + c];
    out[t] = acc;
}

// ---------------- host orchestration ----------------
static inline long long cdivll(long long a, long long b) { return (a + b - 1) / b; }

extern "C" void kernel_launch(void* const* d_in, const int* in_sizes, int n_in,
                              void* d_out, int out_size, void* d_ws, size_t ws_size,
                              hipStream_t stream) {
    (void)in_sizes; (void)n_in; (void)out_size; (void)ws_size;

    const float* x        = (const float*)d_in[0];
    const float* x_ref    = (const float*)d_in[1];
    const int*   eidx     = (const int*)d_in[2];   // (2, EDG)
    const int*   eref     = (const int*)d_in[3];   // (2, EREF)
    const int*   main_idx = (const int*)d_in[4];
    const int*   back_idx = (const int*)d_in[5];
    const float* convs_W0 = (const float*)d_in[7];
    const float* convs_W  = (const float*)d_in[8];
    const float* convs_b  = (const float*)d_in[9];
    const float* ref_W0   = (const float*)d_in[10];
    const float* ref_W    = (const float*)d_in[11];
    const float* ref_b    = (const float*)d_in[12];
    const float* mlp_W0   = (const float*)d_in[13];
    const float* mlp_W1   = (const float*)d_in[14];
    const float* mlp_W2   = (const float*)d_in[15];
    const float* mlp_b2   = (const float*)d_in[16];

    float* out = (float*)d_out;
    float* out_r    = out;                                // (MROWS, 3)
    float* out_back = out + (long long)MROWS * 3;         // (B, WPOOL, H)
    float* out_main = out_back + BG * WPOOL * H;          // (B, WPOOL, H)

    // workspace layout
    char* p = (char*)d_ws;
    auto alloc = [&](size_t bytes) {
        char* r = p;
        p += (bytes + 255) & ~(size_t)255;
        return (void*)r;
    };
    float* deg_m  = (float*)alloc((size_t)NTOT * 4);
    float* nrm_m  = (float*)alloc((size_t)EDG * 4);
    float* T1     = (float*)alloc((size_t)NTOT * H * 4);
    float* T2     = (float*)alloc((size_t)NTOT * H * 4);
    float* act    = (float*)alloc((size_t)NTOT * H * 4);
    float* hbuf   = (float*)alloc((size_t)NTOT * H * 4);
    float* meanb  = (float*)alloc(H * 4);
    float* rstdb  = (float*)alloc(H * 4);
    float* deg_r  = (float*)alloc((size_t)NRES * 4);
    float* nrm_r  = (float*)alloc((size_t)EREF * 4);
    float* rT1    = (float*)alloc((size_t)NRES * H * 4);
    float* rT2    = (float*)alloc((size_t)NRES * H * 4);
    float* ract   = (float*)alloc((size_t)NRES * H * 4);
    float* hrbuf  = (float*)alloc((size_t)NRES * H * 4);
    float* full   = (float*)alloc((size_t)BG * 2 * WPOOL * H * 4);
    half_t* W0t   = (half_t*)alloc((size_t)NINF * NINF * 2);   // transposed f16
    half_t* W1t   = (half_t*)alloc((size_t)NINF * NINF * 2);   // transposed f16
    half_t* resh  = (half_t*)alloc((size_t)MROWS * NINF * 2);
    half_t* r1h   = (half_t*)alloc((size_t)MROWS * NINF * 2);
    half_t* r2h   = resh;  // res no longer needed once GEMM1 is done

    const int TB = 256;

    // ---- graph normalization coefficients ----
    k_zero<<<cdivll(NTOT, TB), TB, 0, stream>>>(deg_m, NTOT);
    k_deg<<<cdivll(EDG, TB), TB, 0, stream>>>(eidx, deg_m, EDG);
    k_norm<<<cdivll(EDG, TB), TB, 0, stream>>>(eidx, eidx + EDG, deg_m, nrm_m, EDG);

    k_zero<<<cdivll(NRES, TB), TB, 0, stream>>>(deg_r, NRES);
    k_deg<<<cdivll(EREF, TB), TB, 0, stream>>>(eref, deg_r, EREF);
    k_norm<<<cdivll(EREF, TB), TB, 0, stream>>>(eref, eref + EREF, deg_r, nrm_r, EREF);

    // ---- main Cheb stack: 3 layers ----
    for (int i = 0; i < 3; ++i) {
        int fin = (i == 0) ? 3 : H;
        const float* t0 = (i == 0) ? x : hbuf;
        const float* W  = (i == 0) ? convs_W0 : (convs_W + (size_t)(i - 1) * 3 * H * H);
        long long nf = (long long)NTOT * fin;

        k_zero<<<cdivll(nf, TB), TB, 0, stream>>>(T1, nf);
        k_prop<<<cdivll((long long)EDG * fin, TB), TB, 0, stream>>>(eidx, eidx + EDG, nrm_m, t0, T1, EDG, fin);
        k_zero<<<cdivll(nf, TB), TB, 0, stream>>>(T2, nf);
        k_prop<<<cdivll((long long)EDG * fin, TB), TB, 0, stream>>>(eidx, eidx + EDG, nrm_m, T1, T2, EDG, fin);
        k_t2<<<cdivll(nf, TB), TB, 0, stream>>>(T2, t0, nf);
        k_cheb<<<cdivll((long long)NTOT * H, TB), TB, 0, stream>>>(t0, T1, T2, W, convs_b + i * H, act, NTOT, fin);
        k_bn_reduce<<<H, 256, 0, stream>>>(act, NTOT, meanb, rstdb);
        k_bn_apply<<<cdivll((long long)NTOT * H, TB), TB, 0, stream>>>(act, meanb, rstdb, hbuf, (long long)NTOT * H);
    }

    // ---- reference Cheb stack: 4 layers ----
    for (int i = 0; i < 4; ++i) {
        int fin = (i == 0) ? 3 : H;
        const float* t0 = (i == 0) ? x_ref : hrbuf;
        const float* W  = (i == 0) ? ref_W0 : (ref_W + (size_t)(i - 1) * 3 * H * H);
        long long nf = (long long)NRES * fin;

        k_zero<<<cdivll(nf, TB), TB, 0, stream>>>(rT1, nf);
        k_prop<<<cdivll((long long)EREF * fin, TB), TB, 0, stream>>>(eref, eref + EREF, nrm_r, t0, rT1, EREF, fin);
        k_zero<<<cdivll(nf, TB), TB, 0, stream>>>(rT2, nf);
        k_prop<<<cdivll((long long)EREF * fin, TB), TB, 0, stream>>>(eref, eref + EREF, nrm_r, rT1, rT2, EREF, fin);
        k_t2<<<cdivll(nf, TB), TB, 0, stream>>>(rT2, t0, nf);
        k_cheb<<<cdivll((long long)NRES * H, TB), TB, 0, stream>>>(t0, rT1, rT2, W, ref_b + i * H, ract, NRES, fin);
        k_bn_reduce<<<H, 256, 0, stream>>>(ract, NRES, meanb, rstdb);
        k_bn_apply<<<cdivll((long long)NRES * H, TB), TB, 0, stream>>>(ract, meanb, rstdb, hrbuf, (long long)NRES * H);
    }

    // ---- adaptive pooling (also fills d_out pooled slices) ----
    k_pool<<<cdivll(BG * WPOOL * H, TB), TB, 0, stream>>>(hbuf, main_idx, NMAIN, full, 0,         out_main);
    k_pool<<<cdivll(BG * WPOOL * H, TB), TB, 0, stream>>>(hbuf, back_idx, NBACK, full, WPOOL * H, out_back);

    // ---- MLP: weight prep (f16 + transpose) + WMMA GEMMs ----
    k_w_f16_t<<<cdivll((long long)NINF * NINF, TB), TB, 0, stream>>>(mlp_W0, W0t, NINF, NINF);
    k_w_f16_t<<<cdivll((long long)NINF * NINF, TB), TB, 0, stream>>>(mlp_W1, W1t, NINF, NINF);
    k_build_res<<<cdivll((long long)MROWS * NINF, TB), TB, 0, stream>>>(hrbuf, full, resh);

    dim3 gemm_grid(NINF / BN, (unsigned)cdivll(MROWS, BM));
    k_gemm_wmma<<<gemm_grid, GT, 0, stream>>>(resh, W0t, r1h, MROWS, NINF, NINF, 1);
    k_gemm_wmma<<<gemm_grid, GT, 0, stream>>>(r1h,  W1t, r2h, MROWS, NINF, NINF, 1);
    k_gemm3<<<cdivll((long long)MROWS * 3, TB), TB, 0, stream>>>(r2h, mlp_W2, mlp_b2, out_r);
}
